// CrystalRGCNEncoder_27522150432769
// MI455X (gfx1250) — matrix operand order repaired and checked by reading, not compiled, not run on hardware
//
#include <hip/hip_runtime.h>
#include <hip/hip_bf16.h>
#include <stdint.h>

// ---------------- problem constants (from reference) ----------------
#define NN      100000
#define NPAD    100096             // NN rounded up to multiple of 128 (GEMM M-tile)
#define EE      1600000
#define GG      128
#define RR      4
#define BB      4
#define D_IN    4
#define HH      64
#define HH2     128
#define LAT     32
#define LN_EPS  1e-5f

// concatenated output widths: [W_r for r in 0..3 | loop_w]
#define OCAT1   ((RR + 1) * HH)    // 320
#define OCAT2   ((RR + 1) * HH2)   // 640
#define KPAD1   32                 // D_IN=4 padded to one WMMA k-step
#define KPAD2   64                 // H=64, two WMMA k-steps

typedef __attribute__((ext_vector_type(16))) __bf16        v16bf;
typedef __attribute__((ext_vector_type(8)))  float         v8f;
typedef __attribute__((ext_vector_type(8)))  unsigned short ushort8;

union FragBF {
    v16bf   v;
    ushort8 h8[2];
};

__device__ __forceinline__ unsigned short f32_to_bf16(float f) {
    unsigned u = __float_as_uint(f);
    unsigned r = 0x7FFFu + ((u >> 16) & 1u);   // round to nearest even
    return (unsigned short)((u + r) >> 16);
}

// ---------------------------------------------------------------
// Build transposed bf16 weight matrix WT[oc][k] (oc = r*O+o, then loop block),
// K zero-padded to Kpad.  W[r] = sum_b comp[r,b]*basis[b]; last O cols = loop_w.
// ---------------------------------------------------------------
__global__ void build_WT(const float* __restrict__ basis, const float* __restrict__ comp,
                         const float* __restrict__ loop_w, unsigned short* __restrict__ WT,
                         int K, int Kpad, int O) {
    int idx  = blockIdx.x * blockDim.x + threadIdx.x;
    int Ocat = (RR + 1) * O;
    if (idx >= Ocat * Kpad) return;
    int oc = idx / Kpad, k = idx - oc * Kpad;
    float v = 0.f;
    if (k < K) {
        if (oc < RR * O) {
            int r = oc / O, o = oc - r * O;
            #pragma unroll
            for (int b = 0; b < BB; ++b)
                v += comp[r * BB + b] * basis[((size_t)b * K + k) * O + o];
        } else {
            int o = oc - RR * O;
            v = loop_w[(size_t)k * O + o];
        }
    }
    WT[(size_t)oc * Kpad + k] = f32_to_bf16(v);
}

// ---------------------------------------------------------------
// t[NPAD][OCAT] = bf16(h[N][KH]) @ Wcat  via v_wmma_f32_16x16x32_bf16.
// Weights staged in LDS (transposed [OCAT][KPAD]) with async global->LDS DMA;
// activations converted to bf16 in LDS. 8 waves/block, 16 rows per wave.
// A fragments hoisted into registers; t is row-padded to NPAD so D stores
// are unconditional (no per-store EXEC masking in the hot loop).
// ---------------------------------------------------------------
template <int KPAD, int OCAT, int KH>
__global__ void __launch_bounds__(256)
rgcn_gemm_wmma(const float* __restrict__ h, const unsigned short* __restrict__ WT,
               float* __restrict__ t, int N) {
    extern __shared__ unsigned short smem[];
    unsigned short* sA = smem;                       // [128][KPAD] bf16
    unsigned short* sB = smem + 128 * KPAD;          // [OCAT][KPAD] bf16 (B^T)

    const int tid = threadIdx.x;
    const int m0  = blockIdx.x * 128;

    // ---- async DMA: weight tile global -> LDS (16B chunks per lane) ----
    {
        constexpr unsigned nChunks = (unsigned)(((size_t)OCAT * KPAD * 2) >> 4);
        unsigned sbBase  = (unsigned)(uintptr_t)(void*)sB;
        unsigned long long gb = (unsigned long long)(uintptr_t)WT;
        for (unsigned c = tid; c < nChunks; c += 256u) {
            unsigned           lds = sbBase + c * 16u;
            unsigned long long ga  = gb + (unsigned long long)c * 16u;
            asm volatile("global_load_async_to_lds_b128 %0, %1, off"
                         :: "v"(lds), "v"(ga) : "memory");
        }
    }

    // ---- stage activations: f32 -> bf16, zero-pad K and tail rows ----
    #pragma unroll
    for (int it = 0; it < (128 * KPAD) / 256; ++it) {
        int idx = tid + it * 256;
        int r = idx / KPAD, k = idx - r * KPAD;
        int gr = m0 + r;
        float val = (gr < N && k < KH) ? h[(size_t)gr * KH + k] : 0.f;
        sA[r * KPAD + k] = f32_to_bf16(val);
    }

    asm volatile("s_wait_asynccnt 0" ::: "memory");
    __syncthreads();

    const int lane = tid & 31;
    const int wave = tid >> 5;
    const int half = lane >> 4;       // 0: low K half, 1: high K half
    const int lrow = lane & 15;
    constexpr int nK     = KPAD / 32;     // 1 or 2
    constexpr int nTiles = OCAT / 16;     // 20 or 40

    // ---- A fragments: loaded once, reused for every column tile ----
    FragBF aF[nK];
    #pragma unroll
    for (int ks = 0; ks < nK; ++ks) {
        // lane (M=lrow); half0 -> K {kb..kb+7, kb+16..kb+23}, half1 -> +8
        const unsigned short* ap = &sA[(wave * 16 + lrow) * KPAD + ks * 32 + half * 8];
        aF[ks].h8[0] = *reinterpret_cast<const ushort8*>(ap);
        aF[ks].h8[1] = *reinterpret_cast<const ushort8*>(ap + 16);
    }

    // per-lane output base: row = m0 + wave*16 + half*8, col = lrow (per tile)
    float* tbase = t + (size_t)(m0 + wave * 16 + half * 8) * OCAT + lrow;

    for (int ct = 0; ct < nTiles; ++ct) {
        v8f acc = {};
        #pragma unroll
        for (int ks = 0; ks < nK; ++ks) {
            FragBF b;
            // lane (N=lrow); half0 -> K {kb..kb+7, kb+16..kb+23}, half1 -> +8
            const unsigned short* bp = &sB[(ct * 16 + lrow) * KPAD + ks * 32 + half * 8];
            b.h8[0] = *reinterpret_cast<const ushort8*>(bp);
            b.h8[1] = *reinterpret_cast<const ushort8*>(bp + 16);
            acc = __builtin_amdgcn_wmma_f32_16x16x32_bf16(
                false, aF[ks].v, false, b.v, (short)0, acc, false, false);
        }
        // D layout: VGPR v -> row v within lane's 8-row group; stores are
        // unconditional (t padded to NPAD rows).
        #pragma unroll
        for (int v = 0; v < 8; ++v)
            tbase[(size_t)v * OCAT + ct * 16] = acc[v];
    }
}

// ---------------------------------------------------------------
// Edge-parallel gather/scatter: agg[dst] += t[etype, src]. float4 gathers,
// f32 atomic scatters (global_atomic_add_f32).  O4 threads per edge.
// ---------------------------------------------------------------
__global__ void __launch_bounds__(256)
rgcn_scatter(const float4* __restrict__ t4, const int* __restrict__ src,
             const int* __restrict__ dst, const int* __restrict__ et,
             float* __restrict__ agg, int E, int O4, int Ocat4, int O) {
    int epb = blockDim.x / O4;
    int e   = blockIdx.x * epb + (int)(threadIdx.x) / O4;
    int c   = (int)(threadIdx.x) & (O4 - 1);
    if (e >= E) return;
    int s = src[e], d = dst[e], r = et[e];
    float4 m = t4[(size_t)s * Ocat4 + r * O4 + c];
    float* ap = agg + (size_t)d * O + c * 4;
    atomicAdd(ap + 0, m.x);
    atomicAdd(ap + 1, m.y);
    atomicAdd(ap + 2, m.z);
    atomicAdd(ap + 3, m.w);
}

// ---------------------------------------------------------------
// Per-node epilogue: out = LN(leaky(agg + loop_term + bias)).  Wave per node.
// ---------------------------------------------------------------
__global__ void rgcn_epilogue(const float* __restrict__ agg, const float* __restrict__ t,
                              const float* __restrict__ bias, const float* __restrict__ lng,
                              const float* __restrict__ lnb, float* __restrict__ out,
                              int N, int O, int Ocat, int loopOff) {
    int wave = threadIdx.x >> 5, lane = threadIdx.x & 31;
    int n = blockIdx.x * (blockDim.x >> 5) + wave;
    if (n >= N) return;
    int per = O >> 5;                         // 2 (H=64) or 4 (H2=128)
    float v[4];
    float s = 0.f;
    for (int i = 0; i < per; ++i) {
        int o = lane + i * 32;
        float x = agg[(size_t)n * O + o] + t[(size_t)n * Ocat + loopOff + o] + bias[o];
        x = (x >= 0.f) ? x : 0.1f * x;
        v[i] = x; s += x;
    }
    for (int off = 16; off; off >>= 1) s += __shfl_xor(s, off, 32);
    float mean = s / (float)O;
    float q = 0.f;
    for (int i = 0; i < per; ++i) { float d = v[i] - mean; q += d * d; }
    for (int off = 16; off; off >>= 1) q += __shfl_xor(q, off, 32);
    float rstd = rsqrtf(q / (float)O + LN_EPS);
    for (int i = 0; i < per; ++i) {
        int o = lane + i * 32;
        out[(size_t)n * O + o] = (v[i] - mean) * rstd * lng[o] + lnb[o];
    }
}

// ---------------------------------------------------------------
// Gate: g[n] = relu(h @ gate_w1 + b1) @ gate_w2 + b2 ; segment max via
// order-preserving uint atomicMax.  Wave per node.
// ---------------------------------------------------------------
__global__ void gate_kernel(const float* __restrict__ h, const float* __restrict__ w1,
                            const float* __restrict__ b1, const float* __restrict__ w2,
                            const float* __restrict__ b2, const int* __restrict__ gid,
                            float* __restrict__ gate, unsigned* __restrict__ gmaxEnc, int N) {
    __shared__ float sh[8][HH2];
    int wave = threadIdx.x >> 5, lane = threadIdx.x & 31;
    int n = blockIdx.x * 8 + wave;
    if (n >= N) return;
    for (int i = lane; i < HH2; i += 32) sh[wave][i] = h[(size_t)n * HH2 + i];
    float a0 = 0.f, a1 = 0.f;
    for (int k = 0; k < HH2; ++k) {
        float hk = sh[wave][k];
        a0 = fmaf(hk, w1[k * HH + lane], a0);
        a1 = fmaf(hk, w1[k * HH + lane + 32], a1);
    }
    a0 = fmaxf(a0 + b1[lane], 0.f);
    a1 = fmaxf(a1 + b1[lane + 32], 0.f);
    float p = a0 * w2[lane] + a1 * w2[lane + 32];
    for (int off = 16; off; off >>= 1) p += __shfl_xor(p, off, 32);
    if (lane == 0) {
        float gv = p + b2[0];
        gate[n] = gv;
        unsigned u = __float_as_uint(gv);
        unsigned enc = (u & 0x80000000u) ? ~u : (u | 0x80000000u);
        atomicMax(&gmaxEnc[gid[n]], enc);
    }
}

__global__ void softmax_e_kernel(const float* __restrict__ gate,
                                 const unsigned* __restrict__ gmaxEnc,
                                 const int* __restrict__ gid, float* __restrict__ e,
                                 float* __restrict__ gsum, int N) {
    int n = blockIdx.x * blockDim.x + threadIdx.x;
    if (n >= N) return;
    unsigned enc = gmaxEnc[gid[n]];
    unsigned u = (enc & 0x80000000u) ? (enc & 0x7FFFFFFFu) : ~enc;
    float ev = expf(gate[n] - __uint_as_float(u));
    e[n] = ev;
    atomicAdd(&gsum[gid[n]], ev);
}

__global__ void pool_kernel(const float* __restrict__ h, const float* __restrict__ e,
                            const float* __restrict__ gsum, const int* __restrict__ gid,
                            float* __restrict__ gemb, int N) {
    int wave = threadIdx.x >> 5, lane = threadIdx.x & 31;
    int n = blockIdx.x * 8 + wave;
    if (n >= N) return;
    int g = gid[n];
    float w = e[n] / gsum[g];
    #pragma unroll
    for (int i = 0; i < 4; ++i) {
        int o = lane + i * 32;
        atomicAdd(&gemb[(size_t)g * HH2 + o], w * h[(size_t)n * HH2 + o]);
    }
}

__global__ void head_kernel(const float* __restrict__ gemb, const float* __restrict__ muw,
                            const float* __restrict__ mub, const float* __restrict__ lvw,
                            const float* __restrict__ lvb, float* __restrict__ mu,
                            float* __restrict__ lv) {
    int g = blockIdx.x;
    int j = threadIdx.x;                       // 64 threads: 32 mu + 32 logvar
    bool isMu = j < 32;
    int  l = j & 31;
    const float* W = isMu ? muw : lvw;
    float acc = 0.f;
    for (int k = 0; k < HH2; ++k) acc = fmaf(gemb[g * HH2 + k], W[k * LAT + l], acc);
    if (isMu) {
        float x = acc + mub[l];
        mu[g * LAT + l] = fminf(fmaxf(x, -5.f), 5.f);
    } else {
        float x = acc + lvb[l];
        lv[g * LAT + l] = fminf(fmaxf(x, -10.f), 10.f);
    }
}

__global__ void zero_f32(float* __restrict__ p, size_t n) {
    size_t i = (size_t)blockIdx.x * blockDim.x + threadIdx.x;
    if (i < n) p[i] = 0.f;
}

__global__ void init_gate_bufs(unsigned* __restrict__ gmaxEnc, float* __restrict__ gsum,
                               float* __restrict__ gemb) {
    int i = blockIdx.x * blockDim.x + threadIdx.x;
    if (i < GG) { gmaxEnc[i] = 0x007FFFFFu; /* enc(-inf) */ gsum[i] = 0.f; }
    if (i < GG * HH2) gemb[i] = 0.f;
}

// ---------------------------------------------------------------
static inline size_t a256(size_t x) { return (x + 255) & ~(size_t)255; }

extern "C" void kernel_launch(void* const* d_in, const int* in_sizes, int n_in,
                              void* d_out, int out_size, void* d_ws, size_t ws_size,
                              hipStream_t stream) {
    (void)in_sizes; (void)n_in; (void)out_size; (void)ws_size;

    const float* node_feats = (const float*)d_in[0];
    const float* basis1     = (const float*)d_in[1];
    const float* comp1      = (const float*)d_in[2];
    const float* loop_w1    = (const float*)d_in[3];
    const float* bias1      = (const float*)d_in[4];
    const float* ln1_g      = (const float*)d_in[5];
    const float* ln1_b      = (const float*)d_in[6];
    const float* basis2     = (const float*)d_in[7];
    const float* comp2      = (const float*)d_in[8];
    const float* loop_w2    = (const float*)d_in[9];
    const float* bias2      = (const float*)d_in[10];
    const float* ln2_g      = (const float*)d_in[11];
    const float* ln2_b      = (const float*)d_in[12];
    const float* gate_w1    = (const float*)d_in[13];
    const float* gate_b1    = (const float*)d_in[14];
    const float* gate_w2    = (const float*)d_in[15];
    const float* gate_b2    = (const float*)d_in[16];
    const float* mu_w       = (const float*)d_in[17];
    const float* mu_b       = (const float*)d_in[18];
    const float* lv_w       = (const float*)d_in[19];
    const float* lv_b       = (const float*)d_in[20];
    const int*   src        = (const int*)d_in[21];
    const int*   dst        = (const int*)d_in[22];
    const int*   etyp       = (const int*)d_in[23];
    const int*   gid        = (const int*)d_in[24];

    // outputs: mu (G*LAT) | logvar (G*LAT) | node_emb (N*H2)
    float* out_mu = (float*)d_out;
    float* out_lv = out_mu + GG * LAT;
    float* h2     = out_lv + GG * LAT;          // node_emb written in place

    // workspace carve-up (t1/t2 row-padded to NPAD for guard-free GEMM stores)
    char* ws = (char*)d_ws;
    size_t off = 0;
    unsigned short* WT1 = (unsigned short*)(ws + off); off += a256((size_t)OCAT1 * KPAD1 * 2);
    unsigned short* WT2 = (unsigned short*)(ws + off); off += a256((size_t)OCAT2 * KPAD2 * 2);
    float* t1   = (float*)(ws + off); off += a256((size_t)NPAD * OCAT1 * 4);
    float* t2   = (float*)(ws + off); off += a256((size_t)NPAD * OCAT2 * 4);
    float* agg1 = (float*)(ws + off); off += a256((size_t)NN * HH * 4);
    float* agg2 = (float*)(ws + off); off += a256((size_t)NN * HH2 * 4);
    float* h1   = (float*)(ws + off); off += a256((size_t)NN * HH * 4);
    float* gate = (float*)(ws + off); off += a256((size_t)NN * 4);
    float* ebuf = (float*)(ws + off); off += a256((size_t)NN * 4);
    unsigned* gmaxEnc = (unsigned*)(ws + off); off += a256((size_t)GG * 4);
    float* gsum = (float*)(ws + off); off += a256((size_t)GG * 4);
    float* gemb = (float*)(ws + off); off += a256((size_t)GG * HH2 * 4);

    // ---- build basis-combined weights (bf16, transposed, K padded) ----
    build_WT<<<(OCAT1 * KPAD1 + 255) / 256, 256, 0, stream>>>(basis1, comp1, loop_w1, WT1, D_IN, KPAD1, HH);
    build_WT<<<(OCAT2 * KPAD2 + 255) / 256, 256, 0, stream>>>(basis2, comp2, loop_w2, WT2, HH, KPAD2, HH2);

    // ---- layer 1 ----
    {
        size_t nz = (size_t)NN * HH;
        zero_f32<<<(unsigned)((nz + 255) / 256), 256, 0, stream>>>(agg1, nz);
        int mblocks = NPAD / 128;
        size_t shm = (size_t)(128 * KPAD1 + OCAT1 * KPAD1) * 2;   // 28.0 KB
        rgcn_gemm_wmma<KPAD1, OCAT1, D_IN><<<mblocks, 256, shm, stream>>>(node_feats, WT1, t1, NN);
        int O4 = HH / 4, epb = 256 / O4;                          // 16 edges/block
        rgcn_scatter<<<(EE + epb - 1) / epb, 256, 0, stream>>>((const float4*)t1, src, dst, etyp,
                                                               agg1, EE, O4, OCAT1 / 4, HH);
        rgcn_epilogue<<<(NN + 7) / 8, 256, 0, stream>>>(agg1, t1, bias1, ln1_g, ln1_b, h1,
                                                        NN, HH, OCAT1, RR * HH);
    }

    // ---- layer 2 ----
    {
        size_t nz = (size_t)NN * HH2;
        zero_f32<<<(unsigned)((nz + 255) / 256), 256, 0, stream>>>(agg2, nz);
        int mblocks = NPAD / 128;
        size_t shm = (size_t)(128 * KPAD2 + OCAT2 * KPAD2) * 2;   // 96 KB (<320 KB WGP LDS)
        rgcn_gemm_wmma<KPAD2, OCAT2, HH><<<mblocks, 256, shm, stream>>>(h1, WT2, t2, NN);
        int O4 = HH2 / 4, epb = 256 / O4;                         // 8 edges/block
        rgcn_scatter<<<(EE + epb - 1) / epb, 256, 0, stream>>>((const float4*)t2, src, dst, etyp,
                                                               agg2, EE, O4, OCAT2 / 4, HH2);
        rgcn_epilogue<<<(NN + 7) / 8, 256, 0, stream>>>(agg2, t2, bias2, ln2_g, ln2_b, h2,
                                                        NN, HH2, OCAT2, RR * HH2);
    }

    // ---- gated softmax pooling + heads ----
    init_gate_bufs<<<(GG * HH2 + 255) / 256, 256, 0, stream>>>(gmaxEnc, gsum, gemb);
    gate_kernel<<<(NN + 7) / 8, 256, 0, stream>>>(h2, gate_w1, gate_b1, gate_w2, gate_b2,
                                                  gid, gate, gmaxEnc, NN);
    softmax_e_kernel<<<(NN + 255) / 256, 256, 0, stream>>>(gate, gmaxEnc, gid, ebuf, gsum, NN);
    pool_kernel<<<(NN + 7) / 8, 256, 0, stream>>>(h2, ebuf, gsum, gid, gemb, NN);
    head_kernel<<<GG, 64, 0, stream>>>(gemb, mu_w, mu_b, lv_w, lv_b, out_mu, out_lv);
}
